// DeformVideo2D_with_FPN_36215164240811
// MI455X (gfx1250) — compile-verified, matrix operand order checked by reading, not run on hardware
//
#include <hip/hip_runtime.h>

// ---------------------------------------------------------------------------
// Deformable-DETR encoder (6 layers) + FPN for MI455X (gfx1250, wave32).
// All GEMMs: bf16 operands (pre-converted, weights transposed to [N][K]),
// f32 accumulation via v_wmma_f32_16x16x32_bf16. Fragments are loaded
// directly from global memory (L2-resident working set), so the K-loop is
// pure b128 loads + WMMA with no LDS staging, no barriers, no conversion.
// ---------------------------------------------------------------------------

#define Bn   4
#define Dn   256
#define LVn  5376           // 64*64 + 32*32 + 16*16
#define MTOK (Bn * LVn)     // 21504

typedef __attribute__((ext_vector_type(16))) __bf16 v16bf;
typedef __attribute__((ext_vector_type(8)))  float  v8f;

union FragAB { v16bf v; uint4 q[2]; };
union FragC  { v8f v; float f[8]; };

__device__ __forceinline__ unsigned short f2bf(float f) {
  unsigned int u = __float_as_uint(f);
  u += 0x7FFFu + ((u >> 16) & 1u);          // round-to-nearest-even
  return (unsigned short)(u >> 16);
}

__device__ __forceinline__ void lvinfo(int q, int& lev, int& start, int& h, int& w) {
  if (q < 4096)      { lev = 0; start = 0;    h = 64; w = 64; }
  else if (q < 5120) { lev = 1; start = 4096; h = 32; w = 32; }
  else               { lev = 2; start = 5120; h = 16; w = 16; }
}

// ---------------------------------------------------------------------------
// WMMA GEMM: C[M,N] = act(A[M,K](bf16) @ Bw[N,K](bf16)^T + bias)
//  - 256 threads = 8 waves (4 along M x 2 along N); block tile 128x128.
//  - Each wave: 32x64 tile = 2x4 wmma fragments, 8 v_wmma per K-step of 32.
//  - SPATIAL: A row remap for 3x3-conv taps with zero border (per-lane,
//    hoisted out of the K-loop).
//  - Optional f32 output C, optional bf16 mirror Cbf, bias/relu/accum/rowmask.
// M must be a multiple of 128.
// ---------------------------------------------------------------------------
template <bool SPATIAL>
__global__ __launch_bounds__(256) void gemm_bf16_k(
    const unsigned short* __restrict__ A,    // [M][K] bf16
    const unsigned short* __restrict__ Bw,   // [N][K] bf16
    const float* __restrict__ bias,
    const unsigned char* __restrict__ rowmask,
    float* __restrict__ C,
    unsigned short* __restrict__ Cbf,
    int M, int N, int K,
    int relu, int accum,
    int Himg, int Wimg, int dy, int dx)
{
  const int tid  = threadIdx.x;
  const int wave = tid >> 5;
  const int lane = tid & 31;
  const int wm   = wave >> 1;               // 0..3 along M
  const int wn   = wave & 1;                // 0..1 along N
  const int m0   = blockIdx.y * 128;
  const int n0   = blockIdx.x * 128;
  const int lrow = lane & 15;
  const int c0   = (lane < 16) ? 0 : 8;     // K sub-chunk per half-wave

  // per-lane A row pointers (2 m-fragments)
  const unsigned short* ap[2];
  bool aok[2];
#pragma unroll
  for (int mf = 0; mf < 2; ++mf) {
    int gm = m0 + wm * 32 + mf * 16 + lrow;
    int srow = gm;
    bool ok = true;
    if (SPATIAL) {
      int hw  = Himg * Wimg;
      int b   = gm / hw;
      int pix = gm - b * hw;
      int y   = pix / Wimg, x = pix - y * Wimg;
      int yy  = y + dy, xx = x + dx;
      ok   = (yy >= 0 && yy < Himg && xx >= 0 && xx < Wimg);
      srow = ok ? (b * hw + yy * Wimg + xx) : 0;
    }
    ap[mf]  = A + (long)srow * K + c0;
    aok[mf] = ok;
  }
  // per-lane B row pointers (4 n-fragments), clamped
  const unsigned short* bp[4];
#pragma unroll
  for (int nf = 0; nf < 4; ++nf) {
    int gn = n0 + wn * 64 + nf * 16 + lrow;
    if (gn >= N) gn = N - 1;
    bp[nf] = Bw + (long)gn * K + c0;
  }

  FragC acc[2][4];
#pragma unroll
  for (int i = 0; i < 2; ++i)
#pragma unroll
    for (int j = 0; j < 4; ++j)
#pragma unroll
      for (int r = 0; r < 8; ++r) acc[i][j].f[r] = 0.f;

  const uint4 zq = make_uint4(0u, 0u, 0u, 0u);
#pragma unroll 2
  for (int k0 = 0; k0 < K; k0 += 32) {
    FragAB a[2], b[4];
#pragma unroll
    for (int mf = 0; mf < 2; ++mf) {
      a[mf].q[0] = *(const uint4*)(ap[mf] + k0);
      a[mf].q[1] = *(const uint4*)(ap[mf] + k0 + 16);
      if (SPATIAL) {
        a[mf].q[0] = aok[mf] ? a[mf].q[0] : zq;
        a[mf].q[1] = aok[mf] ? a[mf].q[1] : zq;
      }
    }
#pragma unroll
    for (int nf = 0; nf < 4; ++nf) {
      b[nf].q[0] = *(const uint4*)(bp[nf] + k0);
      b[nf].q[1] = *(const uint4*)(bp[nf] + k0 + 16);
    }
#pragma unroll
    for (int mf = 0; mf < 2; ++mf)
#pragma unroll
      for (int nf = 0; nf < 4; ++nf)
        acc[mf][nf].v = __builtin_amdgcn_wmma_f32_16x16x32_bf16(
            false, a[mf].v, false, b[nf].v, (short)0, acc[mf][nf].v, false, false);
  }

  // epilogue
  const int hi = (lane < 16) ? 0 : 8;
#pragma unroll
  for (int mf = 0; mf < 2; ++mf)
#pragma unroll
    for (int nf = 0; nf < 4; ++nf) {
      int gn = n0 + wn * 64 + nf * 16 + lrow;
      if (gn >= N) continue;
      float bv = bias ? bias[gn] : 0.f;
#pragma unroll
      for (int r = 0; r < 8; ++r) {
        int gm = m0 + wm * 32 + mf * 16 + r + hi;
        float v = acc[mf][nf].f[r] + bv;
        if (relu) v = fmaxf(v, 0.f);
        long o = (long)gm * N + gn;
        if (accum) v += C[o];
        if (rowmask && rowmask[gm]) v = 0.f;
        if (C)   C[o]   = v;
        if (Cbf) Cbf[o] = f2bf(v);
      }
    }
}

// ---------------------------------------------------------------------------
// Weight conversion: out[n*K + k] = bf16( W[koff + k*kstride + n*nstride] )
// ---------------------------------------------------------------------------
__global__ __launch_bounds__(256) void convw_k(
    const float* __restrict__ W, unsigned short* __restrict__ o,
    int N, int K, int koff, int kstride, int nstride)
{
  long i = (long)blockIdx.x * 256 + threadIdx.x;
  if (i >= (long)N * K) return;
  int k = (int)(i % K);
  int n = (int)(i / K);
  o[i] = f2bf(W[(long)koff + (long)k * kstride + (long)n * nstride]);
}

__global__ void pack_qabias_k(const float* __restrict__ ob,
                              const float* __restrict__ ab,
                              float* __restrict__ dst)
{
  int i = threadIdx.x;
  if (i < 192)      dst[i] = ob[i];
  else if (i < 288) dst[i] = ab[i - 192];
}

// ---------------------------------------------------------------------------
// Setup kernels
// ---------------------------------------------------------------------------
__global__ void compute_vr_k(const unsigned char* __restrict__ pm1,
                             const unsigned char* __restrict__ pm2,
                             const unsigned char* __restrict__ pm3,
                             float* __restrict__ vr)
{
  int t = threadIdx.x;
  if (t >= Bn * 3) return;
  int b = t / 3, l = t % 3;
  int h = (l == 0) ? 64 : (l == 1 ? 32 : 16);
  int w = h;
  const unsigned char* pm = (l == 0) ? pm1 : (l == 1 ? pm2 : pm3);
  int ch = 0, cw = 0;
  for (int y = 0; y < h; ++y) ch += pm[((long)b * h + y) * w] ? 0 : 1;
  for (int x = 0; x < w; ++x) cw += pm[(long)b * h * w + x] ? 0 : 1;
  vr[(b * 3 + l) * 2 + 0] = (float)cw / (float)w;
  vr[(b * 3 + l) * 2 + 1] = (float)ch / (float)h;
}

__global__ __launch_bounds__(256) void build_src_k(
    const float* __restrict__ ms1, const float* __restrict__ ms2, const float* __restrict__ ms3,
    const float* __restrict__ ps1, const float* __restrict__ ps2, const float* __restrict__ ps3,
    const unsigned char* __restrict__ pm1, const unsigned char* __restrict__ pm2,
    const unsigned char* __restrict__ pm3,
    const float* __restrict__ lvl_emb,
    float* __restrict__ src, float* __restrict__ lvlpos,
    unsigned short* __restrict__ src_bf, unsigned short* __restrict__ q_bf,
    unsigned char* __restrict__ mask)
{
  long i = (long)blockIdx.x * 256 + threadIdx.x;
  if (i >= (long)MTOK * Dn) return;
  int c = (int)(i & 255);
  long t = i >> 8;
  int q = (int)(t % LVn);
  int b = (int)(t / LVn);
  int lev, start, h, w;
  lvinfo(q, lev, start, h, w);
  int pix = q - start;
  const float* ms = (lev == 0) ? ms1 : (lev == 1 ? ms2 : ms3);
  const float* ps = (lev == 0) ? ps1 : (lev == 1 ? ps2 : ps3);
  long off = ((long)b * Dn + c) * (h * w) + pix;
  float sv = ms[off];
  float pv = ps[off] + lvl_emb[lev * Dn + c];
  src[i]    = sv;
  lvlpos[i] = pv;
  src_bf[i] = f2bf(sv);
  q_bf[i]   = f2bf(sv + pv);
  if (c == 0) {
    const unsigned char* pm = (lev == 0) ? pm1 : (lev == 1 ? pm2 : pm3);
    mask[(long)b * LVn + q] = pm[(long)b * h * w + pix];
  }
}

__global__ __launch_bounds__(256) void build_ref_k(const float* __restrict__ vr,
                                                   float* __restrict__ refpts)
{
  int i = blockIdx.x * 256 + threadIdx.x;
  if (i >= MTOK) return;
  int q = i % LVn, b = i / LVn;
  int lev, start, h, w;
  lvinfo(q, lev, start, h, w);
  int pix = q - start;
  int gy = pix / w, gx = pix % w;
  float ry = (gy + 0.5f) / (vr[(b * 3 + lev) * 2 + 1] * (float)h);
  float rx = (gx + 0.5f) / (vr[(b * 3 + lev) * 2 + 0] * (float)w);
#pragma unroll
  for (int l2 = 0; l2 < 3; ++l2) {
    refpts[((long)i * 3 + l2) * 2 + 0] = rx * vr[(b * 3 + l2) * 2 + 0];
    refpts[((long)i * 3 + l2) * 2 + 1] = ry * vr[(b * 3 + l2) * 2 + 1];
  }
}

// ---------------------------------------------------------------------------
// Per-layer kernels
// ---------------------------------------------------------------------------
// qaout layout: [tok][288]: cols 0..191 = off (h*24 + (lev*4+p)*2 + xy),
//                           cols 192..287 = aw logits (h*12 + lev*4 + p)
__global__ __launch_bounds__(256) void softmax_loc_k(
    const float* __restrict__ qaout, const float* __restrict__ refpts,
    float* __restrict__ out_aw, float* __restrict__ out_loc)
{
  int i = blockIdx.x * 256 + threadIdx.x;
  if (i >= MTOK * 8) return;
  int h = i & 7;
  long t = i >> 3;
  const float* lg = qaout + t * 288 + 192 + h * 12;
  float mx = lg[0];
#pragma unroll
  for (int j = 1; j < 12; ++j) mx = fmaxf(mx, lg[j]);
  float e[12]; float s = 0.f;
#pragma unroll
  for (int j = 0; j < 12; ++j) { e[j] = expf(lg[j] - mx); s += e[j]; }
  float inv = 1.f / s;
  float* ao = out_aw + (t * 8 + h) * 12;
#pragma unroll
  for (int j = 0; j < 12; ++j) ao[j] = e[j] * inv;

  const float* of = qaout + t * 288 + h * 24;
  const float* rp = refpts + t * 6;
  float* lo = out_loc + (t * 8 + h) * 24;
  const float nrm[3] = {64.f, 32.f, 16.f};
#pragma unroll
  for (int lev = 0; lev < 3; ++lev)
#pragma unroll
    for (int p = 0; p < 4; ++p) {
      int j = lev * 4 + p;
      lo[j * 2 + 0] = rp[lev * 2 + 0] + of[j * 2 + 0] / nrm[lev];
      lo[j * 2 + 1] = rp[lev * 2 + 1] + of[j * 2 + 1] / nrm[lev];
    }
}

__global__ __launch_bounds__(256) void msdeform_k(
    const float* __restrict__ value, const float* __restrict__ locs,
    const float* __restrict__ aws, unsigned short* __restrict__ attn_bf)
{
  __shared__ float sl[8][24];
  __shared__ float sa[8][12];
  long t = blockIdx.x;                 // b*LV + q
  int b = (int)(t / LVn);
  int tid = threadIdx.x;
  for (int i = tid; i < 288; i += 256) {
    if (i < 192) { int hh = i / 24, j = i % 24; sl[hh][j] = locs[(t * 8 + hh) * 24 + j]; }
    else         { int k = i - 192; int hh = k / 12, j = k % 12; sa[hh][j] = aws[(t * 8 + hh) * 12 + j]; }
  }
  __syncthreads();
  int h = tid >> 5, d = tid & 31;
  const int hs[3] = {64, 32, 16}, st[3] = {0, 4096, 5120};
  float acc = 0.f;
  long vb = (long)b * LVn;
#pragma unroll
  for (int lev = 0; lev < 3; ++lev) {
    int H = hs[lev], W = hs[lev], S = st[lev];
#pragma unroll
    for (int p = 0; p < 4; ++p) {
      float lx = sl[h][(lev * 4 + p) * 2 + 0];
      float ly = sl[h][(lev * 4 + p) * 2 + 1];
      float a  = sa[h][lev * 4 + p];
      float x = lx * (float)W - 0.5f;
      float y = ly * (float)H - 0.5f;
      float x0f = floorf(x), y0f = floorf(y);
      int x0 = (int)x0f, y0 = (int)y0f;
      float wx1 = x - x0f, wx0 = 1.f - wx1;
      float wy1 = y - y0f, wy0 = 1.f - wy1;
      float s = 0.f;
#pragma unroll
      for (int dy2 = 0; dy2 < 2; ++dy2)
#pragma unroll
        for (int dx2 = 0; dx2 < 2; ++dx2) {
          int xi = x0 + dx2, yi = y0 + dy2;
          if (xi >= 0 && xi < W && yi >= 0 && yi < H) {
            float wv = (dx2 ? wx1 : wx0) * (dy2 ? wy1 : wy0);
            s += wv * value[(vb + S + yi * W + xi) * 256 + h * 32 + d];
          }
        }
      acc += a * s;
    }
  }
  attn_bf[t * 256 + h * 32 + d] = f2bf(acc);
}

// src = LN(src + delta); also refresh bf16 mirrors: src_bf, q_bf = src+lvlpos
__global__ __launch_bounds__(256) void add_ln_k(
    float* __restrict__ src, const float* __restrict__ delta,
    const float* __restrict__ lvlpos,
    unsigned short* __restrict__ src_bf, unsigned short* __restrict__ q_bf,
    const float* __restrict__ gamma, const float* __restrict__ beta)
{
  __shared__ float red[256];
  long t = blockIdx.x;
  int c = threadIdx.x;
  long o = t * 256 + c;
  float x = src[o] + delta[o];
  red[c] = x;
  __syncthreads();
  for (int s = 128; s > 0; s >>= 1) { if (c < s) red[c] += red[c + s]; __syncthreads(); }
  float mean = red[0] * (1.f / 256.f);
  __syncthreads();
  float d = x - mean;
  red[c] = d * d;
  __syncthreads();
  for (int s = 128; s > 0; s >>= 1) { if (c < s) red[c] += red[c + s]; __syncthreads(); }
  float var = red[0] * (1.f / 256.f);
  float out = d * rsqrtf(var + 1e-5f) * gamma[c] + beta[c];
  src[o]    = out;
  src_bf[o] = f2bf(out);
  q_bf[o]   = f2bf(out + lvlpos[o]);
}

// ---------------------------------------------------------------------------
// FPN kernels
// ---------------------------------------------------------------------------
__global__ __launch_bounds__(256) void chw_to_pm_k(const float* __restrict__ x,
                                                   unsigned short* __restrict__ pm)
{
  long i = (long)blockIdx.x * 256 + threadIdx.x;   // B*16384*256 total
  int c = (int)(i & 255);
  long t = i >> 8;
  int pix = (int)(t & 16383);
  int b = (int)(t >> 14);
  pm[i] = f2bf(x[((long)(b * 256 + c)) * 16384 + pix]);
}

// large_bf = bf16( large_f32 + bilinear_upsample_2x(src level-0 region) )
__global__ __launch_bounds__(256) void upsample_add_k(
    const float* __restrict__ src0, const float* __restrict__ large,
    unsigned short* __restrict__ large_bf)
{
  long i = (long)blockIdx.x * 256 + threadIdx.x;
  int c = (int)(i & 255);
  long t = i >> 8;
  int pix = (int)(t & 16383);
  int b = (int)(t >> 14);
  int y = pix >> 7, x = pix & 127;
  float sy = (float)y * 0.5f - 0.25f;
  float sx = (float)x * 0.5f - 0.25f;
  float y0f = floorf(sy), x0f = floorf(sx);
  int y0 = (int)y0f, x0 = (int)x0f;
  float wy1 = sy - y0f, wy0 = 1.f - wy1;
  float wx1 = sx - x0f, wx0 = 1.f - wx1;
  int yc0 = min(max(y0, 0), 63), yc1 = min(max(y0 + 1, 0), 63);
  int xc0 = min(max(x0, 0), 63), xc1 = min(max(x0 + 1, 0), 63);
  long base = (long)b * LVn;   // level-0 region of src starts at 0
  float v = wy0 * (wx0 * src0[(base + yc0 * 64 + xc0) * 256 + c] +
                   wx1 * src0[(base + yc0 * 64 + xc1) * 256 + c]) +
            wy1 * (wx0 * src0[(base + yc1 * 64 + xc0) * 256 + c] +
                   wx1 * src0[(base + yc1 * 64 + xc1) * 256 + c]);
  large_bf[i] = f2bf(large[i] + v);
}

__global__ __launch_bounds__(256) void groupnorm_k(const float* __restrict__ xpm,
                                                   const float* __restrict__ gs,
                                                   const float* __restrict__ gb,
                                                   float* __restrict__ out)
{
  int b = blockIdx.x >> 5;
  int g = blockIdx.x & 31;
  __shared__ float s1[256], s2[256];
  float sum = 0.f, sq = 0.f;
  for (int idx = threadIdx.x; idx < 16384 * 8; idx += 256) {
    int pix = idx >> 3, cc = idx & 7;
    float v = xpm[((long)b * 16384 + pix) * 256 + g * 8 + cc];
    sum += v; sq += v * v;
  }
  s1[threadIdx.x] = sum; s2[threadIdx.x] = sq;
  __syncthreads();
  for (int s = 128; s > 0; s >>= 1) {
    if (threadIdx.x < s) { s1[threadIdx.x] += s1[threadIdx.x + s]; s2[threadIdx.x] += s2[threadIdx.x + s]; }
    __syncthreads();
  }
  float cnt = 16384.f * 8.f;
  float mean = s1[0] / cnt;
  float var  = s2[0] / cnt - mean * mean;
  float inv  = rsqrtf(var + 1e-5f);
  for (int idx = threadIdx.x; idx < 16384 * 8; idx += 256) {
    int pix = idx >> 3, cc = idx & 7;
    int c = g * 8 + cc;
    float v = xpm[((long)b * 16384 + pix) * 256 + c];
    out[((long)(b * 256 + c)) * 16384 + pix] = (v - mean) * inv * gs[c] + gb[c];
  }
}

__global__ __launch_bounds__(256) void write_feats_k(const float* __restrict__ src,
                                                     float* __restrict__ out)
{
  long i = (long)blockIdx.x * 256 + threadIdx.x;
  if (i >= (long)MTOK * Dn) return;
  int c = (int)(i & 255);
  long t = i >> 8;
  int q = (int)(t % LVn);
  int b = (int)(t / LVn);
  int lev, start, h, w;
  lvinfo(q, lev, start, h, w);
  int pix = q - start;
  const long mb[3] = {16777216L, 20971520L, 22020096L};
  out[mb[lev] + ((long)(b * 256 + c)) * (h * w) + pix] = src[i];
}

// ---------------------------------------------------------------------------
// Host orchestration
// ---------------------------------------------------------------------------
extern "C" void kernel_launch(void* const* d_in, const int* in_sizes, int n_in,
                              void* d_out, int out_size, void* d_ws, size_t ws_size,
                              hipStream_t stream)
{
  (void)in_sizes; (void)n_in; (void)out_size; (void)ws_size;

  const float* ms0 = (const float*)d_in[0];
  const float* ms1 = (const float*)d_in[3];
  const float* ms2 = (const float*)d_in[6];
  const float* ms3 = (const float*)d_in[9];
  const float* ps1 = (const float*)d_in[4];
  const float* ps2 = (const float*)d_in[7];
  const float* ps3 = (const float*)d_in[10];
  const unsigned char* pm1 = (const unsigned char*)d_in[5];
  const unsigned char* pm2 = (const unsigned char*)d_in[8];
  const unsigned char* pm3 = (const unsigned char*)d_in[11];
  // params tree-flattened: fpn{ad_W,ad_b,cv_W,cv_b,gn_b,gn_s}, layers[6]{sorted}, level_embed
  const float* ad_W = (const float*)d_in[12];
  const float* ad_b = (const float*)d_in[13];
  const float* cv_W = (const float*)d_in[14];
  const float* cv_b = (const float*)d_in[15];
  const float* gn_b = (const float*)d_in[16];
  const float* gn_s = (const float*)d_in[17];
  const float* lvl_emb = (const float*)d_in[18 + 6 * 16];
  auto LPP = [&](int L, int k) -> const float* { return (const float*)d_in[18 + L * 16 + k]; };
  // per-layer sorted keys: 0 aw_W, 1 aw_b, 2 ff1_W, 3 ff1_b, 4 ff2_W, 5 ff2_b,
  //                        6 n1_b, 7 n1_s, 8 n2_b, 9 n2_s, 10 off_W, 11 off_b,
  //                        12 out_W, 13 out_b, 14 val_W, 15 val_b

  float* out = (float*)d_out;
  float* ws  = (float*)d_ws;
  const long TOK = (long)MTOK;

  // ---- workspace layout (float units; all region starts 32B aligned) ----
  float* src    = ws;                         // 5,505,024
  float* lvlpos = src    + TOK * 256;         // 5,505,024
  float* value  = lvlpos + TOK * 256;         // 5,505,024
  float* ybuf   = value  + TOK * 256;         // 5,505,024
  float* qaout  = ybuf   + TOK * 256;         // 6,193,152
  float* refpts = qaout  + TOK * 288;         // 129,024
  float* vr     = refpts + TOK * 6;           // 32
  unsigned char* maskf = (unsigned char*)(vr + 32);   // 21,504 B (2048 floats resvd)
  float* bfBase = vr + 32 + 2048;
  unsigned short* src_bf  = (unsigned short*)bfBase;                 // 5,505,024 sh
  unsigned short* q_bf    = src_bf  + TOK * 256;                     // 5,505,024 sh
  unsigned short* attn_bf = q_bf    + TOK * 256;                     // 5,505,024 sh
  unsigned short* ffh_bf  = attn_bf + TOK * 256;                     // 5,505,024 sh (2688x2048)
  unsigned short* a0_bf   = ffh_bf  + TOK * 256;                     // 16,777,216 sh
  unsigned short* lrg_bf  = a0_bf   + 16777216L;                     // 16,777,216 sh
  unsigned short* wtsBase = lrg_bf  + 16777216L;
  // FPN f32 overlays (regions dead by the time FPN runs):
  float* large  = value;                       // 16,777,216 fl over value+ybuf+qaout
  float* convpm = bfBase;                      // 16,777,216 fl over bf16 mirrors+a0_bf
  // weight area (bf16, [N][K]):
  const long LSTR = 1253376L;                  // per-layer shorts
  unsigned short* wad   = wtsBase + 6 * LSTR;              // 65,536
  unsigned short* wconv = wad + 65536L;                    // 9 * 65,536
  float* qabias = (float*)(wconv + 9 * 65536L);            // 288 fl
  auto WL = [&](int L, long off) -> unsigned short* { return wtsBase + L * LSTR + off; };
  const long W_VAL = 0, W_QA = 65536, W_OUT = 139264, W_FF1 = 204800, W_FF2 = 729088;

  const long OUT_LOC = 22282240L, LOCSZ = 4128768L;
  const long OUT_AW  = 47054848L, AWSZ  = 2064384L;

  auto gemm = [&](const unsigned short* A, const unsigned short* Bw, const float* bias,
                  const unsigned char* rowmask, float* C, unsigned short* Cbf,
                  int M, int N, int K, int relu, int accum) {
    dim3 g((N + 127) / 128, M / 128);
    gemm_bf16_k<false><<<g, 256, 0, stream>>>(A, Bw, bias, rowmask, C, Cbf,
                                              M, N, K, relu, accum, 0, 0, 0, 0);
  };
  auto gemmSp = [&](const unsigned short* A, const unsigned short* Bw, const float* bias,
                    float* C, int M, int N, int K, int accum,
                    int Hh, int Ww, int dy, int dx) {
    dim3 g((N + 127) / 128, M / 128);
    gemm_bf16_k<true><<<g, 256, 0, stream>>>(A, Bw, bias, nullptr, C, nullptr,
                                             M, N, K, 0, accum, Hh, Ww, dy, dx);
  };
  auto convw = [&](const float* W, unsigned short* o, int N, int K,
                   int koff, int ks, int ns) {
    long tot = (long)N * K;
    convw_k<<<(unsigned)((tot + 255) / 256), 256, 0, stream>>>(W, o, N, K, koff, ks, ns);
  };

  // ---- weight conversion (f32 -> bf16, transposed to [N][K]) ----
  for (int L = 0; L < 6; ++L) {
    convw(LPP(L, 14), WL(L, W_VAL),            256,  256, 0, 256, 1);   // val_W
    convw(LPP(L, 10), WL(L, W_QA),             192,  256, 0, 192, 1);   // off_W -> qa rows 0..191
    convw(LPP(L, 0),  WL(L, W_QA) + 192 * 256,  96,  256, 0,  96, 1);   // aw_W  -> qa rows 192..287
    convw(LPP(L, 12), WL(L, W_OUT),            256,  256, 0, 256, 1);   // out_W
    convw(LPP(L, 2),  WL(L, W_FF1),           2048,  256, 0, 2048, 1);  // ff1_W
    convw(LPP(L, 4),  WL(L, W_FF2),            256, 2048, 0, 256, 1);   // ff2_W
  }
  convw(ad_W, wad, 256, 256, 0, 1, 256);                                // ad_W [o][i]
  for (int t9 = 0; t9 < 9; ++t9)                                        // cv_W [o][i][ky][kx]
    convw(cv_W, wconv + (long)t9 * 65536, 256, 256, t9, 9, 2304);

  // ---- setup ----
  compute_vr_k<<<1, 32, 0, stream>>>(pm1, pm2, pm3, vr);
  {
    long tot = TOK * 256;
    build_src_k<<<(unsigned)((tot + 255) / 256), 256, 0, stream>>>(
        ms1, ms2, ms3, ps1, ps2, ps3, pm1, pm2, pm3, lvl_emb,
        src, lvlpos, src_bf, q_bf, maskf);
  }
  build_ref_k<<<(MTOK + 255) / 256, 256, 0, stream>>>(vr, refpts);

  // ---- encoder layers ----
  for (int L = 0; L < 6; ++L) {
    // value = src @ val_W + val_b, masked rows -> 0
    gemm(src_bf, WL(L, W_VAL), LPP(L, 15), maskf, value, nullptr,
         MTOK, 256, 256, 0, 0);
    // fused [off | aw] = q @ [off_W | aw_W] + [off_b | aw_b]
    pack_qabias_k<<<1, 288, 0, stream>>>(LPP(L, 11), LPP(L, 1), qabias);
    gemm(q_bf, WL(L, W_QA), qabias, nullptr, qaout, nullptr,
         MTOK, 288, 256, 0, 0);
    // softmax + sampling locations (written straight to outputs)
    float* out_aw  = out + OUT_AW + (long)L * AWSZ;
    float* out_loc = out + OUT_LOC + (long)L * LOCSZ;
    softmax_loc_k<<<(MTOK * 8 + 255) / 256, 256, 0, stream>>>(qaout, refpts,
                                                              out_aw, out_loc);
    // deformable attention gather (writes bf16 attn for the out-proj GEMM)
    msdeform_k<<<MTOK, 256, 0, stream>>>(value, out_loc, out_aw, attn_bf);
    // out projection + residual LN
    gemm(attn_bf, WL(L, W_OUT), LPP(L, 13), nullptr, ybuf, nullptr,
         MTOK, 256, 256, 0, 0);
    add_ln_k<<<MTOK, 256, 0, stream>>>(src, ybuf, lvlpos, src_bf, q_bf,
                                       LPP(L, 7), LPP(L, 6));
    // FFN, chunked rows (8 x 2688) to bound scratch; intermediate kept bf16-only
    for (int ch = 0; ch < 8; ++ch) {
      long ro = (long)ch * 2688;
      gemm(src_bf + ro * 256, WL(L, W_FF1), LPP(L, 3), nullptr, nullptr, ffh_bf,
           2688, 2048, 256, 1, 0);
      gemm(ffh_bf, WL(L, W_FF2), LPP(L, 5), nullptr, ybuf + ro * 256, nullptr,
           2688, 256, 2048, 0, 0);
    }
    add_ln_k<<<MTOK, 256, 0, stream>>>(src, ybuf, lvlpos, src_bf, q_bf,
                                       LPP(L, 9), LPP(L, 8));
  }

  // ---- FPN ----
  chw_to_pm_k<<<16777216 / 256, 256, 0, stream>>>(ms0, a0_bf);
  // adapter 1x1 conv
  gemm(a0_bf, wad, ad_b, nullptr, large, nullptr, 65536, 256, 256, 0, 0);
  // large += upsample2x(feats0); emit bf16 for the conv GEMM
  upsample_add_k<<<16777216 / 256, 256, 0, stream>>>(src, large, lrg_bf);
  // 3x3 conv as 9 accumulating WMMA taps with zero-pad borders
  for (int ky = 0; ky < 3; ++ky)
    for (int kx = 0; kx < 3; ++kx) {
      int first = (ky == 0 && kx == 0);
      gemmSp(lrg_bf, wconv + (long)(ky * 3 + kx) * 65536, first ? cv_b : nullptr,
             convpm, 65536, 256, 256, first ? 0 : 1, 128, 128, ky - 1, kx - 1);
    }
  groupnorm_k<<<128, 256, 0, stream>>>(convpm, gn_s, gn_b, out /* multi[0] */);
  {
    long tot = TOK * 256;
    write_feats_k<<<(unsigned)((tot + 255) / 256), 256, 0, stream>>>(src, out);
  }
}
